// ResNet_30116310680129
// MI455X (gfx1250) — matrix operand order repaired and checked by reading, not compiled, not run on hardware
//
#include <hip/hip_runtime.h>
#include <hip/hip_bf16.h>
#include <math.h>
#include <stdint.h>

// ---------------------------------------------------------------------------
// Capsule EM-routing head for MI455X (gfx1250, wave32, WMMA + TDM).
//
// Pipeline:
//   1) pack_bn      : fold BN into per-channel scale/shift
//   2) pack_weights : w_a(32) + w_pose(512) -> A[576][4608] f16 (pad rows 0)
//   3) im2col       : x -> B[6400][4608] f16 (k-contiguous, 3x3 pad-1)
//   4) pack_Wt      : W[i][j][q][c] -> Wt[i][j][c][q] (float4-friendly)
//   5) gemm_wmma    : C = A*B^T via v_wmma_f32_16x16x32_f16; tiles staged by
//                     the Tensor Data Mover (tensor_load_to_lds, TENSORcnt),
//                     double-buffered in LDS; BN+sigmoid fused in epilogue.
//   6) routing      : per-image EM routing (3 iters) + log-normalized output
// ---------------------------------------------------------------------------

typedef _Float16 half8  __attribute__((ext_vector_type(8)));
typedef _Float16 half16 __attribute__((ext_vector_type(16)));
typedef _Float16 half4  __attribute__((ext_vector_type(4)));
typedef float    f32x8  __attribute__((ext_vector_type(8)));

#define KDIM   4608          // 512 * 9
#define MPAD   576           // 544 padded to 9*64
#define MREAL  544
#define NDIM   6400          // 256 * 25
#define LN2PI_F 1.8378770664093453f

#if __has_builtin(__builtin_amdgcn_tensor_load_to_lds) && \
    __has_builtin(__builtin_amdgcn_s_wait_tensorcnt)
#define USE_TDM 1
#else
#define USE_TDM 0
#endif

// ---------------- 1) BN fold ----------------
__global__ void pack_bn_kernel(const float* __restrict__ ag, const float* __restrict__ ab,
                               const float* __restrict__ am, const float* __restrict__ av,
                               const float* __restrict__ pg, const float* __restrict__ pb,
                               const float* __restrict__ pm, const float* __restrict__ pv,
                               float* __restrict__ scale, float* __restrict__ shift) {
    int m = blockIdx.x * 256 + threadIdx.x;
    if (m >= MREAL) return;
    float g, b, mu, var;
    if (m < 32) { g = ag[m]; b = ab[m]; mu = am[m]; var = av[m]; }
    else        { int c = m - 32; g = pg[c]; b = pb[c]; mu = pm[c]; var = pv[c]; }
    float s = g * rsqrtf(var + 1e-5f);
    scale[m] = s;
    shift[m] = b - mu * s;
}

// ---------------- 2) weight pack (f32 -> f16, fused A matrix) ----------------
__global__ void pack_weights_kernel(const float* __restrict__ wa,
                                    const float* __restrict__ wpose,
                                    _Float16* __restrict__ A) {
    int gid = blockIdx.x * 256 + threadIdx.x;           // MPAD * KDIM / 4 threads
    if (gid >= MPAD * (KDIM / 4)) return;
    int m  = gid / (KDIM / 4);
    int k0 = (gid % (KDIM / 4)) * 4;
    half4 v;
    #pragma unroll
    for (int u = 0; u < 4; ++u) {
        int k = k0 + u;
        float x = 0.0f;
        if (m < 32)        x = wa[(size_t)m * KDIM + k];
        else if (m < 544)  x = wpose[(size_t)(m - 32) * KDIM + k];
        v[u] = (_Float16)x;
    }
    *(half4*)(A + (size_t)m * KDIM + k0) = v;
}

// ---------------- 3) im2col (pad=1 3x3 on 5x5, k-contiguous) ----------------
__global__ void im2col_kernel(const float* __restrict__ x, _Float16* __restrict__ B) {
    int gid = blockIdx.x * 256 + threadIdx.x;           // NDIM * KDIM / 4 threads
    if (gid >= NDIM * (KDIM / 4)) return;
    int n  = gid / (KDIM / 4);
    int k0 = (gid % (KDIM / 4)) * 4;
    int img = n / 25, pix = n % 25;
    int oy = pix / 5, ox = pix % 5;
    half4 v;
    #pragma unroll
    for (int u = 0; u < 4; ++u) {
        int k  = k0 + u;
        int ci = k / 9, r = k % 9;
        int kh = r / 3, kw = r % 3;
        int iy = oy + kh - 1, ix = ox + kw - 1;
        float val = 0.0f;
        if (iy >= 0 && iy < 5 && ix >= 0 && ix < 5)
            val = x[(((size_t)img * 512 + ci) * 5 + iy) * 5 + ix];
        v[u] = (_Float16)val;
    }
    *(half4*)(B + (size_t)n * KDIM + k0) = v;
}

// ---------------- 4) transpose inner 4x4 of routing weight ----------------
__global__ void pack_Wt_kernel(const float* __restrict__ W, float* __restrict__ Wt) {
    int gid = blockIdx.x * 256 + threadIdx.x;           // 800*10*16 threads
    if (gid >= 800 * 10 * 16) return;
    int base = (gid >> 4) << 4;
    int idx  = gid & 15;
    int q = idx >> 2, c = idx & 3;
    Wt[base + c * 4 + q] = W[base + idx];               // [i][j][c][q] = [i][j][q][c]
}

// ---------------- 5) WMMA GEMM + fused BN/sigmoid epilogue ----------------
// LDS pitch: 32 halves of payload + 8 halves pad per row = 40 halves (80 B).
// TDM reproduces this via pad_enable: pad 4 DWORDs after every 16 DWORDs.
#define A_TILE_H (64 * 40)            // halves per A buffer
#define B_TILE_H (256 * 40)           // halves per B buffer

#if USE_TDM
typedef unsigned int u32x4v __attribute__((ext_vector_type(4)));
typedef int          i32x8v __attribute__((ext_vector_type(8)));
typedef int          i32x4v __attribute__((ext_vector_type(4)));

// One 2-D TDM tile load: rows x 32 halves, global row stride KDIM halves,
// LDS destination padded to an 80-byte pitch.
__device__ __forceinline__ void tdm_load_tile(uint64_t gaddr, uint32_t lds_off, int rows) {
    u32x4v g0;
    g0.x = 1u;                                             // count=1, user descriptor
    g0.y = lds_off;                                        // LDS byte address
    g0.z = (uint32_t)gaddr;                                // global_addr[31:0]
    g0.w = ((uint32_t)(gaddr >> 32) & 0x01FFFFFFu)         // global_addr[56:32]
         | (2u << 30);                                     // type = 2 ("image")
    i32x8v g1;
    g1[0] = (1 << 16)                                      // data_size = 2 bytes
          | (1 << 20)                                      // pad_enable
          | (3 << 22)                                      // pad_interval: 16 DWORDs (64 B)
          | (3 << 25);                                     // pad_amount:   4 DWORDs (16 B)
    g1[1] = (int)(4608u << 16);                            // tensor_dim0[15:0] = 4608
    g1[2] = (int)(1024u << 16);                            // tensor_dim1[15:0] = 1024 (no OOB)
    g1[3] = (32 << 16);                                    // tile_dim0 = 32 elements
    g1[4] = rows & 0xFFFF;                                 // tile_dim1 = rows
    g1[5] = 4608;                                          // tensor_dim0_stride[31:0]
    g1[6] = 0;
    g1[7] = 0;
    i32x4v z4 = {0, 0, 0, 0};                              // groups 2/3 unused (2-D tile)
    i32x8v z8 = {0, 0, 0, 0, 0, 0, 0, 0};
    __builtin_amdgcn_tensor_load_to_lds(g0, g1, z4, z4, z8, 0);
}
#endif

__global__ __launch_bounds__(256) void gemm_wmma_kernel(
        const _Float16* __restrict__ A,      // [MPAD][KDIM]
        const _Float16* __restrict__ B,      // [NDIM][KDIM]
        const float* __restrict__ bnScale,   // [544]
        const float* __restrict__ bnShift,   // [544]
        float* __restrict__ convOut) {       // [NDIM][544]
    __shared__ __align__(128) _Float16 As[2 * A_TILE_H];   // double-buffered
    __shared__ __align__(128) _Float16 Bs[2 * B_TILE_H];

    const int tid  = threadIdx.x;
    const int bm   = blockIdx.x;             // 0..8   (M tiles of 64)
    const int bn   = blockIdx.y;             // 0..24  (N tiles of 256)
    const int lane = tid & 31;
    const int wave = tid >> 5;
    const int waveM = wave >> 2;             // 0..1
    const int waveN = wave & 3;              // 0..3
    const int g  = lane >> 4;                // lane half-group
    const int hl = lane & 15;

    f32x8 acc[2][4] = {};
    const int NK = KDIM / 32;

#if USE_TDM
    typedef __attribute__((address_space(3))) _Float16 lds_h;
    const uint32_t asBase = (uint32_t)(uintptr_t)(lds_h*)As;
    const uint32_t bsBase = (uint32_t)(uintptr_t)(lds_h*)Bs;
    const uint64_t aAddr  = (uint64_t)(A + (size_t)(bm * 64) * KDIM);
    const uint64_t bAddr  = (uint64_t)(B + (size_t)(bn * 256) * KDIM);

    if (wave == 0) {                          // prologue: tile 0 -> buffer 0
        tdm_load_tile(aAddr, asBase, 64);
        tdm_load_tile(bAddr, bsBase, 256);
    }
#else
    // synchronous staging fallback: 4 threads per row, 16B per thread
    const int arow = tid >> 2, seg = tid & 3;
    const _Float16* Agp = A + (size_t)(bm * 64 + arow) * KDIM + seg * 8;
    const _Float16* Bgp = B + (size_t)(bn * 256) * KDIM + seg * 8;
#endif

    for (int ks = 0; ks < NK; ++ks) {
        const int buf = ks & 1;
#if USE_TDM
        __syncthreads();   // everyone done reading buf^1 (iteration ks-1)
        if (wave == 0) {
            if (ks + 1 < NK) {
                // prefetch next tile pair into the alternate buffer
                uint64_t koff = (uint64_t)(ks + 1) * 64;   // 32 halves = 64 bytes
                tdm_load_tile(aAddr + koff, asBase + (buf ^ 1) * (A_TILE_H * 2), 64);
                tdm_load_tile(bAddr + koff, bsBase + (buf ^ 1) * (B_TILE_H * 2), 256);
                __builtin_amdgcn_s_wait_tensorcnt(2);      // tile ks complete
            } else {
                __builtin_amdgcn_s_wait_tensorcnt(0);
            }
        }
        __syncthreads();   // publish completed tile ks to all waves
        const _Float16* AsT = As + buf * A_TILE_H;
        const _Float16* BsT = Bs + buf * B_TILE_H;
#else
        const int k0 = ks * 32;
        *(half8*)(As + (tid >> 2) * 40 + (tid & 3) * 8) = *(const half8*)(Agp + k0);
        #pragma unroll
        for (int r = 0; r < 4; ++r) {
            int row = (tid >> 2) + r * 64;
            *(half8*)(Bs + row * 40 + (tid & 3) * 8) =
                *(const half8*)(Bgp + (size_t)row * KDIM + k0);
        }
        if (ks + 1 < NK) {
            __builtin_prefetch(Agp + k0 + 32, 0, 1);
            __builtin_prefetch(Bgp + (size_t)(tid >> 2) * KDIM + k0 + 32, 0, 1);
        }
        __syncthreads();
        const _Float16* AsT = As;
        const _Float16* BsT = Bs;
        (void)buf;
#endif

        // A fragments: lane group g holds K = {8g..8g+7} and {16+8g..16+8g+7}
        half16 af[2];
        #pragma unroll
        for (int fm = 0; fm < 2; ++fm) {
            const _Float16* p = AsT + (waveM * 32 + fm * 16 + hl) * 40;
            half8 lo = *(const half8*)(p + g * 8);
            half8 hi = *(const half8*)(p + 16 + g * 8);
            af[fm] = __builtin_shufflevector(lo, hi, 0,1,2,3,4,5,6,7,8,9,10,11,12,13,14,15);
        }
        // B fragments: lane group g holds K = {8g..8g+7} and {16+8g..16+8g+7} of column N=hl
        half16 bf[4];
        #pragma unroll
        for (int fn = 0; fn < 4; ++fn) {
            const _Float16* p = BsT + (waveN * 64 + fn * 16 + hl) * 40;
            half8 lo = *(const half8*)(p + g * 8);
            half8 hi = *(const half8*)(p + 16 + g * 8);
            bf[fn] = __builtin_shufflevector(lo, hi, 0,1,2,3,4,5,6,7,8,9,10,11,12,13,14,15);
        }

        #pragma unroll
        for (int fm = 0; fm < 2; ++fm)
            #pragma unroll
            for (int fn = 0; fn < 4; ++fn)
                acc[fm][fn] = __builtin_amdgcn_wmma_f32_16x16x32_f16(
                    false, af[fm], false, bf[fn], (short)0, acc[fm][fn], false, false);
#if !USE_TDM
        __syncthreads();
#endif
    }

    // Epilogue: lane holds rows m0..m0+7 (contiguous channels) of column n.
    #pragma unroll
    for (int fm = 0; fm < 2; ++fm) {
        #pragma unroll
        for (int fn = 0; fn < 4; ++fn) {
            int m0 = bm * 64 + waveM * 32 + fm * 16 + 8 * g;
            if (m0 >= MREAL) continue;
            int n = bn * 256 + waveN * 64 + fn * 16 + hl;
            float* dst = convOut + (size_t)n * MREAL + m0;
            bool isAct = (m0 < 32);
            #pragma unroll
            for (int v = 0; v < 8; ++v) {
                float y = acc[fm][fn][v] * bnScale[m0 + v] + bnShift[m0 + v];
                if (isAct) y = 1.0f / (1.0f + __expf(-y));
                dst[v] = y;
            }
        }
    }
}

// ---------------- 6) EM routing (one block per image) ----------------
__global__ __launch_bounds__(256) void routing_kernel(
        const float* __restrict__ convOut,   // [NDIM][544]
        const float* __restrict__ Wt,        // [800][10][4][4] (c-major)
        const float* __restrict__ beta_u,    // [10]
        const float* __restrict__ beta_a,    // [10]
        float* __restrict__ out) {           // [256][10]
    extern __shared__ float smem[];
    float* Psh  = smem;                      // 800*16 pose elements
    float* AUsh = Psh + 800 * 16;            // 800 activations
    float* Rsh  = AUsh + 800;                // 800*10 routing coeffs
    __shared__ float S0sh[16], MUsh[160], SIGsh[160], I2Ssh[160], COSTsh[160];
    __shared__ float AOUTsh[16], C0sh[16];

    const int img = blockIdx.x;
    const int tid = threadIdx.x;
    const float* base = convOut + (size_t)img * 25 * MREAL;

    // capsule i = pix*32 + a : pose channels 32 + a*16 + p, activation channel a
    for (int t = tid; t < 800 * 16; t += 256) {
        int i = t >> 4, p = t & 15;
        int pix = i >> 5, a = i & 31;
        Psh[t] = base[pix * MREAL + 32 + (a << 4) + p];
    }
    for (int i = tid; i < 800; i += 256) {
        int pix = i >> 5, a = i & 31;
        AUsh[i] = base[pix * MREAL + a];
    }
    for (int t = tid; t < 8000; t += 256) Rsh[t] = 1.0f;
    __syncthreads();

    const int j  = tid >> 4;         // class (valid when tid < 160)
    const int p  = tid & 15;         // pose element
    const int rr = (p >> 2) << 2;    // row*4 of 4x4 pose matrix
    const int cc = p & 3;            // col

    for (int it = 0; it < 3; ++it) {
        float lam = 0.01f * (1.0f - powf(0.95f, (float)(it + 1)));

        // ---- pass 1: weighted stats per (j,p) over 800 capsules ----
        float s0 = 0.f, s1 = 0.f, s2 = 0.f;
        if (tid < 160) {
            for (int i = 0; i < 800; ++i) {
                float4 w4 = *(const float4*)(Wt + (((size_t)i * 10 + j) << 4) + (cc << 2));
                float4 p4 = *(const float4*)(Psh + (i << 4) + rr);
                float v  = w4.x * p4.x + w4.y * p4.y + w4.z * p4.z + w4.w * p4.w;
                float rw = Rsh[i * 10 + j] * AUsh[i];
                s0 += rw; s1 += rw * v; s2 += rw * v * v;
            }
            if (p == 0) S0sh[j] = s0;
        }
        __syncthreads();
        if (tid < 160) {
            float rs  = S0sh[j];
            float inv = 1.0f / (rs + 1e-12f);
            float mu  = s1 * inv;
            float sig = fmaxf(s2 * inv - mu * mu, 0.0f) + 1e-12f;
            MUsh[tid]  = mu;
            SIGsh[tid] = sig;
            I2Ssh[tid] = 0.5f / sig;
            COSTsh[tid] = (beta_u[j] + 0.5f * logf(sig)) * rs;
        }
        __syncthreads();
        if (tid < 10) {
            float cs = 0.f, c0 = 0.f;
            #pragma unroll
            for (int pp = 0; pp < 16; ++pp) {
                cs += COSTsh[tid * 16 + pp];
                c0 += logf(SIGsh[tid * 16 + pp] * LN2PI_F);
            }
            float a = 1.0f / (1.0f + expf(-lam * (beta_a[tid] - cs)));
            AOUTsh[tid] = a;
            C0sh[tid]   = -0.5f * c0 + logf(a);     // ln_p constant + ln(a_out)
        }
        __syncthreads();

        // ---- pass 2: softmax update of R ----
        if (it < 2) {
            for (int i = tid; i < 800; i += 256) {
                float lp[10];
                float mx = -3.4e38f;
                #pragma unroll
                for (int jj = 0; jj < 10; ++jj) {
                    float s = C0sh[jj];
                    const float* wb = Wt + (((size_t)i * 10 + jj) << 4);
                    #pragma unroll
                    for (int pr = 0; pr < 4; ++pr) {
                        float4 p4 = *(const float4*)(Psh + (i << 4) + (pr << 2));
                        #pragma unroll
                        for (int pc = 0; pc < 4; ++pc) {
                            float4 w4 = *(const float4*)(wb + (pc << 2));
                            float v = w4.x * p4.x + w4.y * p4.y + w4.z * p4.z + w4.w * p4.w;
                            int pidx = (pr << 2) + pc;
                            float d = v - MUsh[jj * 16 + pidx];
                            s -= d * d * I2Ssh[jj * 16 + pidx];
                        }
                    }
                    lp[jj] = s;
                    mx = fmaxf(mx, s);
                }
                float sum = 0.f;
                #pragma unroll
                for (int jj = 0; jj < 10; ++jj) { lp[jj] = expf(lp[jj] - mx); sum += lp[jj]; }
                float is = 1.0f / sum;
                #pragma unroll
                for (int jj = 0; jj < 10; ++jj) Rsh[i * 10 + jj] = lp[jj] * is;
            }
            __syncthreads();
        }
    }

    if (tid < 10) {
        float s = 0.f;
        #pragma unroll
        for (int jj = 0; jj < 10; ++jj) s += AOUTsh[jj];
        out[img * 10 + tid] = logf(AOUTsh[tid] / s);
    }
}

// ---------------------------------------------------------------------------
extern "C" void kernel_launch(void* const* d_in, const int* in_sizes, int n_in,
                              void* d_out, int out_size, void* d_ws, size_t ws_size,
                              hipStream_t stream) {
    const float* x      = (const float*)d_in[0];
    const float* w_a    = (const float*)d_in[1];
    const float* w_pose = (const float*)d_in[2];
    const float* bn_a_g = (const float*)d_in[3];
    const float* bn_a_b = (const float*)d_in[4];
    const float* bn_a_m = (const float*)d_in[5];
    const float* bn_a_v = (const float*)d_in[6];
    const float* bn_p_g = (const float*)d_in[7];
    const float* bn_p_b = (const float*)d_in[8];
    const float* bn_p_m = (const float*)d_in[9];
    const float* bn_p_v = (const float*)d_in[10];
    const float* W      = (const float*)d_in[11];
    const float* beta_u = (const float*)d_in[12];
    const float* beta_a = (const float*)d_in[13];
    float* out = (float*)d_out;

    // Workspace layout (all offsets 256B aligned); total ~78.8 MB.
    char* ws = (char*)d_ws;
    _Float16* Aglob   = (_Float16*)(ws);                          // 576*4608*2  = 5,308,416
    _Float16* Bglob   = (_Float16*)(ws + 5308416);                // 6400*4608*2 = 58,982,400
    float*    convOut = (float*)(ws + 5308416 + 58982400);        // 6400*544*4  = 13,926,400
    float*    Wt      = (float*)(ws + 5308416 + 58982400 + 13926400);          // 512,000
    float*    bnScale = (float*)(ws + 5308416 + 58982400 + 13926400 + 512000); // 2,176
    float*    bnShift = (float*)(ws + 5308416 + 58982400 + 13926400 + 512000 + 2304);

    pack_bn_kernel<<<3, 256, 0, stream>>>(bn_a_g, bn_a_b, bn_a_m, bn_a_v,
                                          bn_p_g, bn_p_b, bn_p_m, bn_p_v,
                                          bnScale, bnShift);
    pack_weights_kernel<<<(MPAD * (KDIM / 4) + 255) / 256, 256, 0, stream>>>(w_a, w_pose, Aglob);
    im2col_kernel<<<(NDIM * (KDIM / 4) + 255) / 256, 256, 0, stream>>>(x, Bglob);
    pack_Wt_kernel<<<(800 * 10 * 16 + 255) / 256, 256, 0, stream>>>(W, Wt);

    gemm_wmma_kernel<<<dim3(MPAD / 64, NDIM / 256), 256, 0, stream>>>(
        Aglob, Bglob, bnScale, bnShift, convOut);

    size_t routing_lds = (800 * 16 + 800 + 800 * 10) * sizeof(float);  // 86,400 B
    routing_kernel<<<256, 256, routing_lds, stream>>>(convOut, Wt, beta_u, beta_a, out);
}
